// LSTM_W2v_78572131713459
// MI455X (gfx1250) — matrix operand order repaired and checked by reading, not compiled, not run on hardware
//
#include <hip/hip_runtime.h>
#include <cstdint>
#include <cstddef>

// ---------------------------------------------------------------------------
// MI455X (gfx1250) stacked-LSTM implementation.
//
// f16 WMMA (v_wmma_f32_16x16x32_f16) with f32 accumulate. Weights repacked
// once per layer into the per-lane fragment layout from CDNA5 ISA
// so each lane operand is one contiguous v16h (32B) load. Recurrent h-state
// lives in LDS in A-fragment order; the gate phase scatters each new h value
// directly into its WMMA slot, so the next step's GEMM is ds_load_b128 ->
// v_wmma chains. GEMMs are register-blocked 2Mx5N (10 WMMA per 7 loads).
// ---------------------------------------------------------------------------

typedef __attribute__((ext_vector_type(16))) _Float16 v16h;
typedef __attribute__((ext_vector_type(8)))  float    v8f;

#define WAVE 32
#define MB 2
#define NB 5

#define WMMA_F16(a, b, c) \
  __builtin_amdgcn_wmma_f32_16x16x32_f16(false, (a), false, (b), (short)0, (c), false, false)

__device__ __forceinline__ float sigmoidf_(float x) {
  return 1.0f / (1.0f + __expf(-x));
}

// A-matrix (16x32 f16) per-lane layout, inverse mapping:
//   lane = (row&15) + 16*hi ; half j:  j<8 -> K = j + 8*hi ; j>=8 -> K = 16 + (j-8) + 8*hi
__device__ __forceinline__ int afrag_pos(int row, int k, int kcs) {
  int mt   = row >> 4;
  int kc   = k >> 5;
  int koff = k & 31;
  int hi, j;
  if (koff < 16) { hi = koff >> 3;        j = koff & 7; }
  else           { hi = (koff - 16) >> 3; j = ((koff - 16) & 7) + 8; }
  return ((mt * kcs + kc) * WAVE + ((row & 15) | (hi << 4))) * 16 + j;
}

// ---------------------------------------------------------------------------
// Repack f32 weight matrix W[N][K] (row-major) into f16 B-fragment tiles:
// frag[ntile][kc][lane][16].  B layout: col N=lane&15, hi=lane>>4 ;
// j<8 -> K=j+8*hi ; j>=8 -> K=16+(j-8)+8*hi  (within a 32-deep K chunk).
// ---------------------------------------------------------------------------
__global__ void k_conv_bfrag(const float* __restrict__ W, _Float16* __restrict__ frag,
                             int N, int K, int ntiles, int kcs) {
  size_t total = (size_t)ntiles * kcs * WAVE * 16;
  size_t id = (size_t)blockIdx.x * blockDim.x + threadIdx.x;
  if (id >= total) return;
  int    j    = (int)(id & 15);
  int    lane = (int)((id >> 4) & 31);
  size_t tt   = id >> 9;
  int kc = (int)(tt % (size_t)kcs);
  int nt = (int)(tt / (size_t)kcs);
  int n = nt * 16 + (lane & 15);
  int k = kc * 32 + ((j < 8) ? (j + ((lane >> 4) << 3)) : (8 + j + ((lane >> 4) << 3)));
  float v = (n < N && k < K) ? W[(size_t)n * K + k] : 0.0f;
  frag[id] = (_Float16)v;
}

// ---------------------------------------------------------------------------
// Repack a T-chunk of activations X[B][T][D] (f32, b-major) into f16
// A-fragment tiles. Logical GEMM row r = t_local*128 + b  (8192 rows/chunk).
// ---------------------------------------------------------------------------
__global__ void k_conv_afrag(const float* __restrict__ X, _Float16* __restrict__ frag,
                             int T, int D, int t0, int kcs) {
  size_t total = (size_t)512 * kcs * WAVE * 16;   // 512 M-tiles per chunk
  size_t id = (size_t)blockIdx.x * blockDim.x + threadIdx.x;
  if (id >= total) return;
  int    j    = (int)(id & 15);
  int    lane = (int)((id >> 4) & 31);
  size_t tt   = id >> 9;
  int kc = (int)(tt % (size_t)kcs);
  int mt = (int)(tt / (size_t)kcs);
  int row  = mt * 16 + (lane & 15);
  int hi   = lane >> 4;
  int koff = (j < 8) ? (j + (hi << 3)) : (8 + j + (hi << 3));
  int k = kc * 32 + koff;
  int t = t0 + (row >> 7);
  int b = row & 127;
  float v = (k < D) ? X[((size_t)b * T + t) * D + k] : 0.0f;
  frag[id] = (_Float16)v;
}

__global__ void k_bias_sum(const float* __restrict__ a, const float* __restrict__ b,
                           float* __restrict__ o, int n) {
  int i = blockIdx.x * blockDim.x + threadIdx.x;
  if (i < n) o[i] = a[i] + b[i];
}

// ---------------------------------------------------------------------------
// Batched input-projection GEMM: out[r,n] = A@B + bias[n].
// Register-blocked: each wave owns a 2Mx5N block of 16x16 tiles.
// Per K-chunk: 2 A loads + 5 B loads feed 10 v_wmma (vs 2 loads / 1 wmma).
// ---------------------------------------------------------------------------
__global__ void k_gemm_pre(const _Float16* __restrict__ afrag,
                           const _Float16* __restrict__ bfrag,
                           const float* __restrict__ bias,
                           float* __restrict__ out,
                           int mtiles, int ntiles, int kcs, int N4) {
  const int mg = mtiles / MB, ng = ntiles / NB;   // exact: mtiles even, ntiles % 5 == 0
  int gwave = (int)(((size_t)blockIdx.x * blockDim.x + threadIdx.x) >> 5);
  int lane  = threadIdx.x & 31;
  if (gwave >= mg * ng) return;                   // wave-uniform exit (EXEC stays full)
  int mi = gwave / ng, ni = gwave - mi * ng;

  const v16h* ap0 = (const v16h*)afrag + ((size_t)(mi * MB + 0) * kcs) * WAVE + lane;
  const v16h* ap1 = (const v16h*)afrag + ((size_t)(mi * MB + 1) * kcs) * WAVE + lane;
  const v16h* bp  = (const v16h*)bfrag + ((size_t)(ni * NB) * kcs) * WAVE + lane;

  v8f acc[MB][NB] = {};
  for (int kc = 0; kc < kcs; ++kc) {
    if (kc + 1 < kcs) {                           // global_prefetch_b8 next K-chunk
      __builtin_prefetch((const void*)(ap0 + (size_t)(kc + 1) * WAVE), 0, 1);
      __builtin_prefetch((const void*)(bp + (size_t)(kc + 1) * WAVE), 0, 1);
    }
    v16h a0 = ap0[(size_t)kc * WAVE];
    v16h a1 = ap1[(size_t)kc * WAVE];
#pragma unroll
    for (int n = 0; n < NB; ++n) {
      v16h b = bp[((size_t)n * kcs + kc) * WAVE];
      acc[0][n] = WMMA_F16(a0, b, acc[0][n]);
      acc[1][n] = WMMA_F16(a1, b, acc[1][n]);
    }
  }
#pragma unroll
  for (int m = 0; m < MB; ++m) {
    int row0 = (mi * MB + m) * 16 + ((lane >> 4) << 3);
#pragma unroll
    for (int n = 0; n < NB; ++n) {
      int   ncol = (ni * NB + n) * 16 + (lane & 15);
      float bv   = bias[ncol];
      float* orow = out + (size_t)row0 * N4 + ncol;
#pragma unroll
      for (int r = 0; r < 8; ++r) orow[(size_t)r * N4] = acc[m][n][r] + bv;
    }
  }
}

// ---------------------------------------------------------------------------
// Sequential recurrence over one T-chunk. Single 1024-thread workgroup
// (32 waves on one WGP). h lives in dynamic LDS as A-fragments (80KB for
// H=300, legal on the 320KB WGP). c stays in an L2-hot global buffer.
// Phase-1 GEMM is register-blocked 2Mx5N like the batched GEMM.
// ---------------------------------------------------------------------------
__global__ void __launch_bounds__(1024) k_lstm_rec(
    const float* __restrict__ pre,       // [tcount*128][4H], row = t_local*128 + b
    const _Float16* __restrict__ whhfrag,
    float* __restrict__ gbuf,            // [128][4H] gate scratch (L2-resident)
    float* __restrict__ y,               // [B][T][H] layer output stream
    float* __restrict__ hstate,          // [128*H]  canonical h (chunk carry)
    float* __restrict__ cstate,          // [128*H]  canonical c (chunk carry)
    int H, int kcs, int t0, int tcount, int T) {
  extern __shared__ _Float16 hfrag[];    // [8][kcs][32][16]
  const int N4     = 4 * H;
  const int ntiles = N4 >> 4;
  const int tid    = threadIdx.x;
  const int lane   = tid & 31;
  const int wv     = tid >> 5;
  const int nthr   = blockDim.x;
  const int nwv    = nthr >> 5;

  // Zero fragment (covers K padding), then load carried h-state into slots.
  const int hfragN = 8 * kcs * WAVE * 16;
  for (int i = tid; i < hfragN; i += nthr) hfrag[i] = (_Float16)0.0f;
  __syncthreads();
  for (int i = tid; i < 128 * H; i += nthr) {
    int b = i / H, k = i - b * H;
    hfrag[afrag_pos(b, k, kcs)] = (_Float16)hstate[i];
  }
  __syncthreads();

  const int mg = 8 / MB;                 // 4 M-groups (B=128 -> 8 M-tiles)
  const int ng = ntiles / NB;            // 15 (H=300) or 5 (H=100)
  const int units = mg * ng;

  for (int tl = 0; tl < tcount; ++tl) {
    const float* pre_t = pre + (size_t)tl * 128 * N4;

    // Phase 1: g = pre + h @ Whh^T  (2Mx5N WMMA blocks strided across waves)
    for (int u = wv; u < units; u += nwv) {
      int mi = u / ng, ni = u - mi * ng;
      const v16h* ap0 = (const v16h*)hfrag + ((size_t)(mi * MB + 0) * kcs) * WAVE + lane;
      const v16h* ap1 = (const v16h*)hfrag + ((size_t)(mi * MB + 1) * kcs) * WAVE + lane;
      const v16h* bp  = (const v16h*)whhfrag + ((size_t)(ni * NB) * kcs) * WAVE + lane;
      v8f acc[MB][NB] = {};
      for (int kc = 0; kc < kcs; ++kc) {
        v16h a0 = ap0[(size_t)kc * WAVE];          // ds_load_b128 x2
        v16h a1 = ap1[(size_t)kc * WAVE];
#pragma unroll
        for (int n = 0; n < NB; ++n) {
          v16h b = bp[((size_t)n * kcs + kc) * WAVE];
          acc[0][n] = WMMA_F16(a0, b, acc[0][n]);
          acc[1][n] = WMMA_F16(a1, b, acc[1][n]);
        }
      }
#pragma unroll
      for (int m = 0; m < MB; ++m) {
        int row0 = (mi * MB + m) * 16 + ((lane >> 4) << 3);
#pragma unroll
        for (int n = 0; n < NB; ++n) {
          int ncol = (ni * NB + n) * 16 + (lane & 15);
          const float* pr = pre_t + (size_t)row0 * N4 + ncol;
          float*       gr = gbuf  + (size_t)row0 * N4 + ncol;
#pragma unroll
          for (int r = 0; r < 8; ++r) gr[(size_t)r * N4] = acc[m][n][r] + pr[(size_t)r * N4];
        }
      }
    }
    __syncthreads();

    // Phase 2: gate nonlinearities, c/h update; new h written into its
    // A-fragment slot for the next step's WMMA.
    int tg = t0 + tl;
    for (int i = tid; i < 128 * H; i += nthr) {
      int b = i / H, n = i - b * H;
      const float* g  = gbuf + (size_t)b * N4;
      float ig = sigmoidf_(g[n]);
      float fg = sigmoidf_(g[H + n]);
      float gg = tanhf(g[2 * H + n]);
      float og = sigmoidf_(g[3 * H + n]);
      float cv = fg * cstate[i] + ig * gg;
      cstate[i] = cv;
      float hv = og * tanhf(cv);
      hfrag[afrag_pos(b, n, kcs)] = (_Float16)hv;
      y[((size_t)b * T + tg) * H + n] = hv;
      if (tl == tcount - 1) hstate[i] = hv;      // chunk carry-out
    }
    __syncthreads();
  }
}

__global__ void k_add(float* __restrict__ a, const float* __restrict__ b, size_t n) {
  size_t i = (size_t)blockIdx.x * blockDim.x + threadIdx.x;
  if (i < n) a[i] += b[i];
}

// Prediction head: feat=[cT,hT] -> Linear(200,50) -> BN -> LeakyReLU ->
// Linear(50,2) -> softmax.  One thread per batch row (tiny).
__global__ void k_head(const float* __restrict__ hT, const float* __restrict__ cT,
                       const float* __restrict__ W1, const float* __restrict__ b1,
                       const float* __restrict__ gamma, const float* __restrict__ beta,
                       const float* __restrict__ rmean, const float* __restrict__ rvar,
                       const float* __restrict__ W2, const float* __restrict__ b2,
                       float* __restrict__ out) {
  int b = blockIdx.x * blockDim.x + threadIdx.x;
  if (b >= 128) return;
  float z1[50];
  for (int o = 0; o < 50; ++o) {
    float s = b1[o];
    const float* w = W1 + o * 200;
    for (int k = 0; k < 100; ++k) s += cT[b * 100 + k] * w[k];
    for (int k = 0; k < 100; ++k) s += hT[b * 100 + k] * w[100 + k];
    s = (s - rmean[o]) * rsqrtf(rvar[o] + 1e-5f) * gamma[o] + beta[o];
    z1[o] = (s >= 0.0f) ? s : 0.1f * s;
  }
  float z2[2];
  for (int o = 0; o < 2; ++o) {
    float s = b2[o];
    for (int k = 0; k < 50; ++k) s += z1[k] * W2[o * 50 + k];
    z2[o] = s;
  }
  float m  = fmaxf(z2[0], z2[1]);
  float e0 = __expf(z2[0] - m), e1 = __expf(z2[1] - m);
  float inv = 1.0f / (e0 + e1);
  out[b * 2 + 0] = e0 * inv;
  out[b * 2 + 1] = e1 * inv;
}

// ---------------------------------------------------------------------------
// Orchestration. Input flattening (depth-first over the setup_inputs dict):
//   d_in[0] = x ; d_in[1 + s*16 + l*4 + {Wih,Whh,bih,bhh}] ; last 8 = head.
// ---------------------------------------------------------------------------
extern "C" void kernel_launch(void* const* d_in, const int* in_sizes, int n_in,
                              void* d_out, int out_size, void* d_ws, size_t ws_size,
                              hipStream_t stream) {
  (void)in_sizes; (void)out_size; (void)ws_size;
  const int B = 128, T = 512, D0 = 300;
  const int TCH = 64, NCH = T / TCH;       // 8 T-chunks of 64 steps

  char* ws = (char*)d_ws;
  size_t off = 0;
  auto alloc = [&](size_t bytes) -> char* {
    char* p = ws + off; off += (bytes + 255) & ~(size_t)255; return p;
  };
  float*    buf_res = (float*)alloc((size_t)B * T * 300 * 4);   // residual stream
  float*    buf_a   = (float*)alloc((size_t)B * T * 300 * 4);   // layer ping
  float*    buf_b   = (float*)alloc((size_t)B * T * 300 * 4);   // layer pong
  float*    pre_c   = (float*)alloc((size_t)TCH * B * 1200 * 4);// pre-activation chunk
  _Float16* afrag   = (_Float16*)alloc((size_t)512 * 10 * WAVE * 16 * 2);
  _Float16* wihf    = (_Float16*)alloc((size_t)75 * 10 * WAVE * 16 * 2);
  _Float16* whhf    = (_Float16*)alloc((size_t)75 * 10 * WAVE * 16 * 2);
  float*    biasb   = (float*)alloc(1200 * 4);
  float*    gbuf    = (float*)alloc((size_t)128 * 1200 * 4);
  float*    hstate  = (float*)alloc((size_t)128 * 300 * 4);
  float*    cstate  = (float*)alloc((size_t)128 * 300 * 4);

  // Dynamic LDS up to 80KB (WGP has 320KB).
  hipFuncSetAttribute(reinterpret_cast<const void*>(&k_lstm_rec),
                      hipFuncAttributeMaxDynamicSharedMemorySize,
                      8 * 10 * WAVE * 16 * 2);

  hipMemcpyAsync(buf_res, d_in[0], (size_t)B * T * D0 * 4,
                 hipMemcpyDeviceToDevice, stream);

  for (int s = 0; s < 5; ++s) {
    const int H = (s == 4) ? 100 : 300;
    const int N4 = 4 * H, ntiles = N4 / 16;
    const int kcs_h = (H + 31) / 32;
    const float* layer_in = buf_res;
    float* yout = nullptr;
    for (int l = 0; l < 4; ++l) {
      const int Din = (l == 0) ? 300 : H;
      const int kcs_in = (Din + 31) / 32;
      const int base = 1 + s * 16 + l * 4;
      const float* Wih = (const float*)d_in[base + 0];
      const float* Whh = (const float*)d_in[base + 1];
      const float* bih = (const float*)d_in[base + 2];
      const float* bhh = (const float*)d_in[base + 3];
      yout = (l & 1) ? buf_b : buf_a;

      { size_t tot = (size_t)ntiles * kcs_in * 512;
        k_conv_bfrag<<<(unsigned)((tot + 255) / 256), 256, 0, stream>>>(
            Wih, wihf, N4, Din, ntiles, kcs_in); }
      { size_t tot = (size_t)ntiles * kcs_h * 512;
        k_conv_bfrag<<<(unsigned)((tot + 255) / 256), 256, 0, stream>>>(
            Whh, whhf, N4, H, ntiles, kcs_h); }
      k_bias_sum<<<(N4 + 255) / 256, 256, 0, stream>>>(bih, bhh, biasb, N4);
      hipMemsetAsync(hstate, 0, (size_t)128 * H * 4, stream);
      hipMemsetAsync(cstate, 0, (size_t)128 * H * 4, stream);

      for (int c = 0; c < NCH; ++c) {
        { size_t tot = (size_t)512 * kcs_in * 512;
          k_conv_afrag<<<(unsigned)((tot + 255) / 256), 256, 0, stream>>>(
              layer_in, afrag, T, Din, c * TCH, kcs_in); }
        { int units = (512 / MB) * (ntiles / NB);   // 8 waves / 256-thread block
          k_gemm_pre<<<(units + 7) / 8, 256, 0, stream>>>(
              afrag, wihf, biasb, pre_c, 512, ntiles, kcs_in, N4); }
        { int lds = 8 * kcs_h * WAVE * 16 * 2;
          k_lstm_rec<<<1, 1024, lds, stream>>>(
              pre_c, whhf, gbuf, yout, hstate, cstate, H, kcs_h, c * TCH, TCH, T); }
      }
      layer_in = yout;
    }
    if (s < 4)
      k_add<<<(unsigned)(((size_t)B * T * 300 + 255) / 256), 256, 0, stream>>>(
          buf_res, yout, (size_t)B * T * 300);
  }

  const int hb = n_in - 8;
  k_head<<<1, 128, 0, stream>>>(hstate, cstate,
      (const float*)d_in[hb + 0], (const float*)d_in[hb + 1],
      (const float*)d_in[hb + 2], (const float*)d_in[hb + 3],
      (const float*)d_in[hb + 4], (const float*)d_in[hb + 5],
      (const float*)d_in[hb + 6], (const float*)d_in[hb + 7],
      (float*)d_out);
}
